// rs_CF_10780367913202
// MI455X (gfx1250) — matrix operand order repaired and checked by reading, not compiled, not run on hardware
//
#include <hip/hip_runtime.h>
#include <hip/hip_fp16.h>

typedef __attribute__((ext_vector_type(16))) _Float16 v16h;
typedef __attribute__((ext_vector_type(8)))  _Float16 v8h;
typedef __attribute__((ext_vector_type(8)))  float    v8f;

#define N_USERS 8192
#define N_ITEMS 4096
#define TOPK    400
#define EPSV    1e-5f

// ---------------------------------------------------------------------------
// 1) per-row L2 norm:  rnorm[i] = sqrt(sum_j R[i,j]^2) + eps
// ---------------------------------------------------------------------------
__global__ __launch_bounds__(256) void rownorm_kernel(const float* __restrict__ R,
                                                      float* __restrict__ rnorm) {
  const int row = blockIdx.x;
  const float* p = R + (size_t)row * N_ITEMS;
  float s = 0.f;
  for (int j = threadIdx.x; j < N_ITEMS; j += 256) { float v = p[j]; s += v * v; }
  __shared__ float red[256];
  red[threadIdx.x] = s;
  __syncthreads();
  for (int off = 128; off > 0; off >>= 1) {
    if (threadIdx.x < off) red[threadIdx.x] += red[threadIdx.x + off];
    __syncthreads();
  }
  if (threadIdx.x == 0) rnorm[row] = sqrtf(red[0]) + EPSV;
}

// ---------------------------------------------------------------------------
// 2) per-item mean: imean[j] = sum_i R[i,j] / (count(R[i,j]>0) + eps)
// ---------------------------------------------------------------------------
__global__ __launch_bounds__(256) void colstats_kernel(const float* __restrict__ R,
                                                       float* __restrict__ imean) {
  const int j = blockIdx.x * 256 + threadIdx.x;
  float s = 0.f, c = 0.f;
  for (int i = 0; i < N_USERS; ++i) {
    float v = R[(size_t)i * N_ITEMS + j];
    s += v;
    c += (v > 0.f) ? 1.f : 0.f;
  }
  imean[j] = s / (c + EPSV);
}

// ---------------------------------------------------------------------------
// 3) convert: Rn16 = f16(R/rnorm) row-major; RT16 = f16(R)^T; MT16 = f16(R>0)^T
// ---------------------------------------------------------------------------
__global__ __launch_bounds__(256) void convert_kernel(const float* __restrict__ R,
                                                      const float* __restrict__ rnorm,
                                                      _Float16* __restrict__ Rn16,
                                                      _Float16* __restrict__ RT16,
                                                      _Float16* __restrict__ MT16) {
  __shared__ _Float16 tR[32 * 33];
  __shared__ _Float16 tM[32 * 33];
  const int tx = threadIdx.x & 31, ty = threadIdx.x >> 5;
  const int j0 = blockIdx.x * 32, i0 = blockIdx.y * 32;
  for (int yy = 0; yy < 4; ++yy) {
    int iofs = ty + 8 * yy;
    int row = i0 + iofs, col = j0 + tx;
    float v = R[(size_t)row * N_ITEMS + col];
    Rn16[(size_t)row * N_ITEMS + col] = (_Float16)(v / rnorm[row]);
    tR[iofs * 33 + tx] = (_Float16)v;
    tM[iofs * 33 + tx] = (v > 0.f) ? (_Float16)1.f : (_Float16)0.f;
  }
  __syncthreads();
  for (int yy = 0; yy < 4; ++yy) {
    int jofs = ty + 8 * yy;
    int trow = j0 + jofs, tcol = i0 + tx;
    RT16[(size_t)trow * N_USERS + tcol] = tR[tx * 33 + jofs];
    MT16[(size_t)trow * N_USERS + tcol] = tM[tx * 33 + jofs];
  }
}

// ---------------------------------------------------------------------------
// 4) NT WMMA GEMM:  C[M,N] (f32) = A[M,K] (f16) * B[N,K]^T (f16)
//    block = 256 thr (8 waves), tile 128x128, K-slab 64, double-buffered LDS
//    filled with GLOBAL_LOAD_ASYNC_TO_LDS_B128 (ASYNCcnt), so the whole
//    global fetch of slab i+1 overlaps the 16 v_wmma of slab i.
//    LDS rows padded to 72 halves (144 B) for aligned b128 fragment reads.
// ---------------------------------------------------------------------------
#define BK        64
#define LDSTRIDE  72                 // halves per LDS row (64 + 8 pad)
#define BUFH      (128 * LDSTRIDE)   // halves per buffer per matrix

__global__ __launch_bounds__(256) void gemm_nt_wmma(const _Float16* __restrict__ A,
                                                    const _Float16* __restrict__ B,
                                                    float* __restrict__ C,
                                                    int Kdim, int Ndim) {
  __shared__ _Float16 As[2 * BUFH];  // 36 KB
  __shared__ _Float16 Bs[2 * BUFH];  // 36 KB
  const int tid   = threadIdx.x;
  const int lane  = tid & 31;        // wave32
  const int wave  = tid >> 5;
  const int wrow  = wave >> 2;       // 0..1
  const int wcol  = wave & 3;        // 0..3
  const int rowBase = blockIdx.y * 128;
  const int colBase = blockIdx.x * 128;
  const int lhalf = (lane < 16) ? 0 : 1;
  const int l16   = lane & 15;
  const unsigned asBase = (unsigned)(uintptr_t)&As[0];
  const unsigned bsBase = (unsigned)(uintptr_t)&Bs[0];

  v8f zero = {};
  v8f acc[4][2];
  for (int i = 0; i < 4; ++i)
    for (int j = 0; j < 2; ++j) acc[i][j] = zero;

  // Issue one 128xBK slab of A and B into LDS buffer `buf` (async, no VGPR data).
  // 8 async-b128 instructions per wave -> ASYNCcnt += 8.
  auto issue_slab = [&](int buf, int kk0) {
    for (int c = 0; c < 4; ++c) {
      int chunk = tid + 256 * c;            // 0..1023
      int r  = chunk >> 3;                  // 0..127
      int c8 = (chunk & 7) * 8;             // 0..56 halves
      unsigned loff = (unsigned)((buf * BUFH + r * LDSTRIDE + c8) * 2);
      unsigned la = asBase + loff;
      unsigned lb = bsBase + loff;
      unsigned long long ga =
          (unsigned long long)(uintptr_t)(A + (size_t)(rowBase + r) * Kdim + kk0 + c8);
      unsigned long long gb =
          (unsigned long long)(uintptr_t)(B + (size_t)(colBase + r) * Kdim + kk0 + c8);
      asm volatile("global_load_async_to_lds_b128 %0, %1, off" :: "v"(la), "v"(ga) : "memory");
      asm volatile("global_load_async_to_lds_b128 %0, %1, off" :: "v"(lb), "v"(gb) : "memory");
    }
  };

  issue_slab(0, 0);  // prologue: ASYNCcnt = 8

  for (int k0 = 0; k0 < Kdim; k0 += BK) {
    const int buf = (k0 >> 6) & 1;
    __syncthreads();  // all waves done computing from buf^1 -> safe to overwrite
    if (k0 + BK < Kdim) {
      issue_slab(buf ^ 1, k0 + BK);                       // ASYNCcnt: 8 -> 16
      asm volatile("s_wait_asynccnt 0x8" ::: "memory");   // slab `buf` landed (in-order)
    } else {
      asm volatile("s_wait_asynccnt 0x0" ::: "memory");
    }
    __syncthreads();  // slab `buf` visible to every wave

    const _Float16* Ab = As + buf * BUFH;
    const _Float16* Bb = Bs + buf * BUFH;
    for (int kk = 0; kk < 2; ++kk) {
      const int kofs = kk * 32;
      // B fragments: 32x16, lane<16 holds K 0..15, lane>=16 holds K 16..31
      v16h bfrag[2];
      for (int sc = 0; sc < 2; ++sc) {
        int bn = wcol * 32 + sc * 16 + l16;
        v8h lo = *(const v8h*)(&Bb[bn * LDSTRIDE + kofs + lhalf * 16]);
        v8h hi = *(const v8h*)(&Bb[bn * LDSTRIDE + kofs + lhalf * 16 + 8]);
        for (int e = 0; e < 8; ++e) { bfrag[sc][e] = lo[e]; bfrag[sc][e + 8] = hi[e]; }
      }
      // A fragments: 16x32, lane<16 holds K 0..15, lane>=16 holds K 16..31
      for (int sr = 0; sr < 4; ++sr) {
        int ar = wrow * 64 + sr * 16 + l16;
        v8h lo = *(const v8h*)(&Ab[ar * LDSTRIDE + kofs + lhalf * 16]);
        v8h hi = *(const v8h*)(&Ab[ar * LDSTRIDE + kofs + lhalf * 16 + 8]);
        v16h af;
        for (int e = 0; e < 8; ++e) { af[e] = lo[e]; af[e + 8] = hi[e]; }
        acc[sr][0] = __builtin_amdgcn_wmma_f32_16x16x32_f16(
            false, af, false, bfrag[0], (short)0, acc[sr][0], false, false);
        acc[sr][1] = __builtin_amdgcn_wmma_f32_16x16x32_f16(
            false, af, false, bfrag[1], (short)0, acc[sr][1], false, false);
      }
    }
  }

  // C/D layout: VGPR v, lane<16 -> M=v, N=lane; lane>=16 -> M=v+8, N=lane-16
  for (int sr = 0; sr < 4; ++sr)
    for (int sc = 0; sc < 2; ++sc) {
      int cb = colBase + wcol * 32 + sc * 16 + l16;
      int rb = rowBase + wrow * 64 + sr * 16 + (lhalf ? 8 : 0);
      for (int v = 0; v < 8; ++v)
        C[(size_t)(rb + v) * Ndim + cb] = acc[sr][sc][v];
    }
}

// ---------------------------------------------------------------------------
// 5) exact top-K threshold per row via MSB-first radix select on
//    order-preserving uint transform of f32; emit f16 masked row of Ds.
// ---------------------------------------------------------------------------
__global__ __launch_bounds__(256) void topk_kernel(const float* __restrict__ D,
                                                   _Float16* __restrict__ Ds) {
  __shared__ unsigned int sord[N_USERS];   // 32 KB
  __shared__ unsigned int hist[256];
  __shared__ unsigned int sh_prefix, sh_rank;
  const int row = blockIdx.x;
  const int tid = threadIdx.x;
  const float* Drow = D + (size_t)row * N_USERS;

  for (int j = tid; j < N_USERS; j += 256) {
    unsigned int u = __float_as_uint(Drow[j]);
    sord[j] = u ^ ((u >> 31) ? 0xFFFFFFFFu : 0x80000000u);  // total order
  }
  if (tid == 0) { sh_prefix = 0u; sh_rank = TOPK - 1; }
  __syncthreads();

  for (int b = 3; b >= 0; --b) {
    hist[tid] = 0u;
    __syncthreads();
    unsigned int prefix = sh_prefix;
    for (int j = tid; j < N_USERS; j += 256) {
      unsigned int v = sord[j];
      bool match = (b == 3) ? true : ((v >> ((b + 1) * 8)) == prefix);
      if (match) atomicAdd(&hist[(v >> (b * 8)) & 0xFFu], 1u);
    }
    __syncthreads();
    if (tid == 0) {
      unsigned int rank = sh_rank, cum = 0u, bin = 0u;
      for (int t = 255; t >= 0; --t) {
        unsigned int h = hist[t];
        if (cum + h > rank) { bin = (unsigned int)t; sh_rank = rank - cum; break; }
        cum += h;
      }
      sh_prefix = (prefix << 8) | bin;
    }
    __syncthreads();
  }

  const unsigned int T = sh_prefix;  // exact ordered-uint value of K-th largest
  for (int j = tid; j < N_USERS; j += 256) {
    unsigned int v = sord[j];
    _Float16 outv = (_Float16)0.f;
    if (v >= T) {
      unsigned int u = (v & 0x80000000u) ? (v ^ 0x80000000u) : ~v;  // inverse map
      outv = (_Float16)__uint_as_float(u);
    }
    Ds[(size_t)row * N_USERS + j] = outv;
  }
}

// ---------------------------------------------------------------------------
// 6) finish: P = num/(den+eps); fallback to item mean where P <= 0 (in place)
// ---------------------------------------------------------------------------
__global__ __launch_bounds__(256) void final_kernel(float* __restrict__ P,
                                                    const float* __restrict__ Pden,
                                                    const float* __restrict__ imean) {
  size_t idx = (size_t)blockIdx.x * 256 + threadIdx.x;
  int j = (int)(idx & (N_ITEMS - 1));
  float p = P[idx] / (Pden[idx] + EPSV);
  P[idx] = (p > 0.f) ? p : imean[j];
}

// ---------------------------------------------------------------------------
extern "C" void kernel_launch(void* const* d_in, const int* in_sizes, int n_in,
                              void* d_out, int out_size, void* d_ws, size_t ws_size,
                              hipStream_t stream) {
  const float* R = (const float*)d_in[0];
  float* out = (float*)d_out;
  char* ws = (char*)d_ws;

  size_t o = 0;
  _Float16* Rn16 = (_Float16*)(ws + o); o += (size_t)N_USERS * N_ITEMS * 2;  //  64 MB
  _Float16* RT16 = (_Float16*)(ws + o); o += (size_t)N_ITEMS * N_USERS * 2;  // 128 MB
  _Float16* MT16 = (_Float16*)(ws + o); o += (size_t)N_ITEMS * N_USERS * 2;  // 192 MB
  float*    Dm   = (float*)   (ws + o); o += (size_t)N_USERS * N_USERS * 4;  // 448 MB
  _Float16* Ds16 = (_Float16*)(ws + o); o += (size_t)N_USERS * N_USERS * 2;  // 576 MB
  float*    Pden = (float*)   (ws + o); o += (size_t)N_USERS * N_ITEMS * 4;  // 704 MB
  float*    rnorm= (float*)   (ws + o); o += (size_t)N_USERS * 4;
  float*    imean= (float*)   (ws + o);

  rownorm_kernel <<<N_USERS, 256, 0, stream>>>(R, rnorm);
  colstats_kernel<<<N_ITEMS / 256, 256, 0, stream>>>(R, imean);
  convert_kernel <<<dim3(N_ITEMS / 32, N_USERS / 32), 256, 0, stream>>>(R, rnorm, Rn16, RT16, MT16);
  // D = Rn * Rn^T   (M=8192, N=8192, K=4096)
  gemm_nt_wmma   <<<dim3(N_USERS / 128, N_USERS / 128), 256, 0, stream>>>(Rn16, Rn16, Dm, N_ITEMS, N_USERS);
  topk_kernel    <<<N_USERS, 256, 0, stream>>>(Dm, Ds16);
  // Pnum = Ds * R   (B = R^T stored [N,K]) -> straight into d_out
  gemm_nt_wmma   <<<dim3(N_ITEMS / 128, N_USERS / 128), 256, 0, stream>>>(Ds16, RT16, out, N_USERS, N_ITEMS);
  // Pden = Ds * mask
  gemm_nt_wmma   <<<dim3(N_ITEMS / 128, N_USERS / 128), 256, 0, stream>>>(Ds16, MT16, Pden, N_USERS, N_ITEMS);
  final_kernel   <<<((size_t)N_USERS * N_ITEMS) / 256, 256, 0, stream>>>(out, Pden, imean);
}
